// GCNN_49331994362309
// MI455X (gfx1250) — matrix operand order repaired and checked by reading, not compiled, not run on hardware
//
#include <hip/hip_runtime.h>
#include <hip/hip_bf16.h>

typedef __attribute__((ext_vector_type(16))) _Float16 v16h;
typedef __attribute__((ext_vector_type(8)))  _Float16 v8h;
typedef __attribute__((ext_vector_type(8)))  float    v8f;

#define NNODE 1024
#define KNB   32

// ---------------- Stage A: graph conv helpers (negligible FLOPs) ----------------

// xej[n][k] = mean_j e[n][j][k]  ; e: (N,32,32)
__global__ void mean_e_kernel(const float* __restrict__ e, float* __restrict__ xej) {
    int idx = blockIdx.x * blockDim.x + threadIdx.x;        // N*32
    int nrow = idx >> 5, k = idx & 31;
    float s = 0.f;
    const float* base = e + nrow * (KNB * 32) + k;
    #pragma unroll
    for (int j = 0; j < KNB; ++j) s += base[j * 32];
    xej[idx] = s * (1.f / 32.f);
}

// xnj[n][k] = mean_j x[ij[n][j]][k]
__global__ void gather_mean_kernel(const float* __restrict__ x, const int* __restrict__ ij,
                                   float* __restrict__ xnj, int F) {
    int idx = blockIdx.x * blockDim.x + threadIdx.x;        // N*F
    int nrow = idx / F, k = idx - nrow * F;
    const int* nb = ij + nrow * KNB;
    float s = 0.f;
    #pragma unroll
    for (int j = 0; j < KNB; ++j) s += x[nb[j] * F + k];
    xnj[idx] = s * (1.f / 32.f);
}

// out[n][c] = relu( x[n]·Wc[c] + xnj[n]·Wn[c] + xej[n]·We[c] )
__global__ void conv_kernel(const float* __restrict__ x, const float* __restrict__ xnj,
                            const float* __restrict__ xej,
                            const float* __restrict__ Wc, const float* __restrict__ Wn,
                            const float* __restrict__ We,
                            float* __restrict__ outp, int Fin, int Fout) {
    int idx = blockIdx.x * blockDim.x + threadIdx.x;        // N*Fout
    int nrow = idx / Fout, c = idx - nrow * Fout;
    const float* xr  = x   + nrow * Fin;
    const float* xjr = xnj + nrow * Fin;
    const float* wc  = Wc  + c * Fin;
    const float* wn  = Wn  + c * Fin;
    float s = 0.f;
    for (int k = 0; k < Fin; ++k) s = fmaf(xr[k], wc[k], fmaf(xjr[k], wn[k], s));
    const float* xe = xej + nrow * 32;
    const float* we = We + c * 32;
    #pragma unroll
    for (int k = 0; k < 32; ++k) s = fmaf(xe[k], we[k], s);
    outp[idx] = fmaxf(s, 0.f);
}

// u[i][c] = 0.5 * sum_k A1[c][k] * x3[i][k]   (A1: 128x64, x3: Nx64) -> f16
__global__ void proj_kernel(const float* __restrict__ x3, const float* __restrict__ A1,
                            _Float16* __restrict__ u) {
    int idx = blockIdx.x * blockDim.x + threadIdx.x;        // N*128
    int i = idx >> 7, c = idx & 127;
    const float* xr = x3 + i * 64;
    const float* ar = A1 + c * 64;
    float s = 0.f;
    #pragma unroll
    for (int k = 0; k < 64; ++k) s = fmaf(ar[k], xr[k], s);
    u[idx] = (_Float16)(0.5f * s);
}

__global__ void cvt_f16_kernel(const float* __restrict__ src, _Float16* __restrict__ dst) {
    int idx = blockIdx.x * blockDim.x + threadIdx.x;
    dst[idx] = (_Float16)src[idx];
}

// ---------------- Stage C: pairwise dense, all-WMMA (compute-bound, ~17 GFLOP) ----------------
// Per 16-pair tile: h1 = relu(ul_i + vr_j) built in B layout; h2 = A2 @ h1 via 16
// v_wmma_f32_16x16x32_f16 (A2 register-resident, identical for all waves); the final
// 2x64 layer is ALSO a WMMA: the h2 accumulators in C/D layout are bit-for-bit the
// A-operand layout of h2^T, so out^T = h2^T @ A3pad^T needs only an in-lane f32->f16
// convert (+relu) and 2 more WMMAs against a 16-VGPR constant B operand. Lanes n<2
// then hold 8 contiguous outputs each -> b128 stores. No shuffles, no 64-reg A3 table;
// wave stays under 256 VGPRs (no s_set_vgpr_msb).
__global__ __launch_bounds__(256) void pair_dense_kernel(
    const _Float16* __restrict__ ul,   // (1024,128)
    const _Float16* __restrict__ vr,   // (1024,128)
    const _Float16* __restrict__ A2h,  // (64,128)
    const float*    __restrict__ A3,   // (2,64)
    float*          __restrict__ out)  // (2, 1024*1024)
{
    const int lane   = threadIdx.x & 31;
    const int wave   = threadIdx.x >> 5;
    const int waveId = blockIdx.x * 8 + wave;        // 8192 waves
    const int i      = waveId >> 3;                  // xl row
    const int jbase  = (waveId & 7) << 3;            // first of 8 j-tiles
    const int n      = lane & 15;
    const int hi     = lane >> 4;

    // ---- register-resident A2 operands: A[mt*4+k] (A layout: e <-> K=hi*16+e)
    v16h A[16];
    #pragma unroll
    for (int mt = 0; mt < 4; ++mt) {
        #pragma unroll
        for (int k = 0; k < 4; ++k) {
            const _Float16* arow = A2h + (mt * 16 + n) * 128 + k * 32 + hi * 16;
            v8h lo = *(const v8h*)(arow);
            v8h hh = *(const v8h*)(arow + 8);
            v16h a;
            #pragma unroll
            for (int e = 0; e < 8; ++e) { a[e] = lo[e]; a[e + 8] = hh[e]; }
            A[mt * 4 + k] = a;
        }
    }

    // ---- constant B operand for the output layer: A3^T zero-padded to 64x16, f16,
    //      B layout: chunk c2, elem e<8 <-> d = c2*32 + hi*8 + e, e>=8 <-> d = c2*32 + 16 + hi*8 + (e-8), column = n (only n<2 non-zero)
    v16h B3[2];
    #pragma unroll
    for (int c2 = 0; c2 < 2; ++c2) {
        v16h b = {};
        if (n < 2) {
            const float* a3p = A3 + n * 64 + c2 * 32 + hi * 8;
            #pragma unroll
            for (int e = 0; e < 16; ++e) b[e] = (_Float16)a3p[e + (e & 8)];
        }
        B3[c2] = b;
    }

    const _Float16* ulrow = ul + i * 128 + hi * 16;

    __builtin_prefetch(vr + ((jbase + 0) * 16 + n) * 128, 0, 0);
    __builtin_prefetch(vr + ((jbase + 1) * 16 + n) * 128, 0, 0);

    #pragma unroll 1
    for (int t = 0; t < 8; ++t) {
        const int j0 = (jbase + t) << 4;

        // prefetch two tiles ahead (harmless past range)
        __builtin_prefetch(vr + ((j0 + 32) + n) * 128, 0, 0);

        // ---- B operand: h1 = relu(ul_i + vr_j) in B layout (e <-> c = k*32 + hi*16 + e)
        const _Float16* vrrow = vr + (j0 + n) * 128 + hi * 16;
        v16h hb[4];
        #pragma unroll
        for (int k = 0; k < 4; ++k) {
            v16h uv = *(const v16h*)(ulrow + k * 32);   // wave-constant, L0-hot
            v16h b  = *(const v16h*)(vrrow + k * 32);
            v16h s  = uv + b;
            #pragma unroll
            for (int e = 0; e < 16; ++e)
                s[e] = (s[e] > (_Float16)0.f) ? s[e] : (_Float16)0.f;
            hb[k] = s;
        }

        // ---- h2 = relu(A2 @ h1), then out^T = h2^T @ A3pad^T (accumulated over 2 chunks)
        v8f d2 = {};
        #pragma unroll
        for (int c2 = 0; c2 < 2; ++c2) {
            v8f acc0 = {}, acc1 = {};
            #pragma unroll
            for (int k = 0; k < 4; ++k) {
                acc0 = __builtin_amdgcn_wmma_f32_16x16x32_f16(
                    false, A[(2 * c2 + 0) * 4 + k], false, hb[k], (short)0, acc0, false, false);
                acc1 = __builtin_amdgcn_wmma_f32_16x16x32_f16(
                    false, A[(2 * c2 + 1) * 4 + k], false, hb[k], (short)0, acc1, false, false);
            }
            // D layout of acc0/acc1 == A layout of h2^T for K-chunk c2: just relu+cvt in-lane
            v16h a2;
            #pragma unroll
            for (int e = 0; e < 8; ++e) {
                a2[e]     = (_Float16)fmaxf(acc0[e], 0.f);
                a2[e + 8] = (_Float16)fmaxf(acc1[e], 0.f);
            }
            d2 = __builtin_amdgcn_wmma_f32_16x16x32_f16(
                false, a2, false, B3[c2], (short)0, d2, false, false);
        }

        // d2: lane(n,hi) VGPR r = out_pre[o=n][pair=hi*8+r]; only n<2 meaningful
        if (n < 2) {
            float* op = out + n * 1048576 + i * 1024 + j0 + hi * 8;
            #pragma unroll
            for (int r = 0; r < 8; ++r) op[r] = fmaxf(d2[r], 0.f);
        }
    }
}

// ---------------- driver ----------------

extern "C" void kernel_launch(void* const* d_in, const int* in_sizes, int n_in,
                              void* d_out, int out_size, void* d_ws, size_t ws_size,
                              hipStream_t stream) {
    const float* xnr = (const float*)d_in[0];
    const float* xer = (const float*)d_in[1];
    const int*   ijr = (const int*)  d_in[2];
    const float* xnl = (const float*)d_in[3];
    const float* xel = (const float*)d_in[4];
    const int*   ijl = (const int*)  d_in[5];
    const float* Wc1 = (const float*)d_in[6];
    const float* Wn1 = (const float*)d_in[7];
    const float* We1 = (const float*)d_in[8];
    const float* Wc2 = (const float*)d_in[9];
    const float* Wn2 = (const float*)d_in[10];
    const float* We2 = (const float*)d_in[11];
    const float* Wc3 = (const float*)d_in[12];
    const float* Wn3 = (const float*)d_in[13];
    const float* We3 = (const float*)d_in[14];
    const float* A1  = (const float*)d_in[15];
    const float* A2  = (const float*)d_in[16];
    const float* A3  = (const float*)d_in[17];
    float* out = (float*)d_out;

    char* ws = (char*)d_ws;
    float*    xej = (float*)(ws + 0);            // 128 KB
    float*    xnj = (float*)(ws + 131072);       // 512 KB
    float*    x1  = (float*)(ws + 655360);       // 512 KB
    float*    x2  = (float*)(ws + 1179648);      // 512 KB
    float*    x3l = (float*)(ws + 1703936);      // 256 KB
    float*    x3r = (float*)(ws + 1966080);      // 256 KB
    _Float16* ul  = (_Float16*)(ws + 2228224);   // 256 KB
    _Float16* vrp = (_Float16*)(ws + 2490368);   // 256 KB
    _Float16* A2h = (_Float16*)(ws + 2752512);   // 16 KB

    const int N = NNODE, T = 256;

    auto run_side = [&](const float* xn0, const float* e, const int* ij,
                        float* x3, _Float16* u) {
        mean_e_kernel     <<<(N * 32) / T, T, 0, stream>>>(e, xej);
        gather_mean_kernel<<<(N * 64) / T, T, 0, stream>>>(xn0, ij, xnj, 64);
        conv_kernel       <<<(N * 128) / T, T, 0, stream>>>(xn0, xnj, xej, Wc1, Wn1, We1, x1, 64, 128);
        gather_mean_kernel<<<(N * 128) / T, T, 0, stream>>>(x1, ij, xnj, 128);
        conv_kernel       <<<(N * 128) / T, T, 0, stream>>>(x1, xnj, xej, Wc2, Wn2, We2, x2, 128, 128);
        gather_mean_kernel<<<(N * 128) / T, T, 0, stream>>>(x2, ij, xnj, 128);
        conv_kernel       <<<(N * 64) / T, T, 0, stream>>>(x2, xnj, xej, Wc3, Wn3, We3, x3, 128, 64);
        proj_kernel       <<<(N * 128) / T, T, 0, stream>>>(x3, A1, u);
    };

    run_side(xnl, xel, ijl, x3l, ul);
    run_side(xnr, xer, ijr, x3r, vrp);
    cvt_f16_kernel<<<(64 * 128) / T, T, 0, stream>>>(A2, A2h);

    // 8192 waves x 8 tiles = 65536 tiles of 16 pairs; 8 waves/block -> 1024 blocks
    pair_dense_kernel<<<1024, 256, 0, stream>>>(ul, vrp, A2h, A3, out);
}
